// ProjectiveAttention_66434554135242
// MI455X (gfx1250) — matrix operand order repaired and checked by reading, not compiled, not run on hardware
//
#include <hip/hip_runtime.h>
#include <hip/hip_bf16.h>
#include <stdint.h>

// ---------------------------------------------------------------------------
// ProjectiveAttention on MI455X (gfx1250, wave32, WMMA).
//   B=8, S=2048, D=800.  ~129 GFLOP vs ~110MB HBM traffic -> compute bound.
//   All 3 GEMMs on v_wmma_f32_16x16x32_f16 (f16 in, f32 accumulate).
//   Register tiles: 32x32 (proj), 32x64 (scores), 64x32 (out) per wave32
//   -> 4/8/8 WMMAs per K-step, ~32-43 flop/byte from L2.
// Workspace layout (bytes):
//   h16  @ 0          : 26,214,400  (f16 copy of h, row-major [B,S,D])
//   Wt16 @ 26,214,400 :  1,280,000  (f16 W transposed [e][d])
//   Q16  @ 27,494,400 : 26,214,400  (f16 h@W; later ALIASED as ht16 [B,D,S])
//   attn @ 53,708,800 : 67,108,864  (f16 scores -> softmax in place)
// Total ~120.8 MB (fits 192MB L2 -> score round-trip stays on-chip).
// ---------------------------------------------------------------------------

typedef _Float16 half_t;
typedef __attribute__((ext_vector_type(16))) half_t v16h;
typedef __attribute__((ext_vector_type(8)))  half_t v8h;
typedef __attribute__((ext_vector_type(2)))  half_t v2h;
typedef __attribute__((ext_vector_type(8)))  float  v8f;

#define BB 8
#define SS 2048
#define DD 800

__device__ __forceinline__ v8f wmma_f16(v16h a, v16h b, v8f c) {
  // v_wmma_f32_16x16x32_f16  (neg_a, A, neg_b, B, c_mod, C, reuse_a, reuse_b)
  return __builtin_amdgcn_wmma_f32_16x16x32_f16(false, a, false, b, (short)0, c,
                                                false, false);
}

// A fragment: 16x32 f16, row-major source, `base` points at (row0, k0),
// ld = row stride in halves. Layout: lane = 16*hi + m;
// VGPR j : K = 16*hi + 2j,2j+1 (matches load_b_frag).
// Per lane this is two contiguous 16B chunks -> compiler emits 2x b128.
__device__ __forceinline__ v16h load_a_frag(const half_t* __restrict__ base,
                                            int ld, int lane) {
  const int m = lane & 15, hi = lane >> 4;
  const half_t* rp = base + (size_t)m * ld;
  v16h a;
#pragma unroll
  for (int j = 0; j < 8; ++j) {
    const int koff = hi * 16 + 2 * j;
    v2h p = *(const v2h*)(rp + koff);
    a[2 * j] = p.x;
    a[2 * j + 1] = p.y;
  }
  return a;
}

// B fragment: 32x16 f16 where source is "N-major": row n of `base` holds the
// K-contiguous values of output column n (i.e. B[k][n] = base[n*ld + k]).
// Layout used here (same k order as load_a_frag): lane = 16*kgrp + n ; half j = B[16*kgrp + j][n].
// -> one 32-byte contiguous vector load per lane (2x b128).
__device__ __forceinline__ v16h load_b_frag(const half_t* __restrict__ base,
                                            int ld, int lane) {
  const int n = lane & 15, kb = (lane >> 4) * 16;
  return *(const v16h*)(base + (size_t)n * ld + kb);
}

// C layout: VGPR r -> M = r + 8*hi, N = lane&15.
__device__ __forceinline__ void store_c_f16(v8f c, half_t* __restrict__ dst,
                                            int ld, int lane) {
  const int n = lane & 15, hi = lane >> 4;
#pragma unroll
  for (int r = 0; r < 8; ++r) dst[(size_t)(r + 8 * hi) * ld + n] = (half_t)c[r];
}
__device__ __forceinline__ void store_c_f32(v8f c, float* __restrict__ dst,
                                            int ld, int lane) {
  const int n = lane & 15, hi = lane >> 4;
#pragma unroll
  for (int r = 0; r < 8; ++r) dst[(size_t)(r + 8 * hi) * ld + n] = c[r];
}

// ---- stage 0: conversions -------------------------------------------------
__global__ void k_convert_h(const float* __restrict__ h,
                            half_t* __restrict__ h16) {
  size_t i = (size_t)blockIdx.x * blockDim.x + threadIdx.x;  // exact grid
  h16[i] = (half_t)h[i];
}

__global__ void k_convert_w(const float* __restrict__ W,
                            half_t* __restrict__ Wt) {
  int i = blockIdx.x * blockDim.x + threadIdx.x;  // D*D exact
  int d = i / DD, e = i % DD;
  Wt[(size_t)e * DD + d] = (half_t)W[i];
}

// transpose h16 [B,S,D] -> ht16 [B,D,S] (runs after Q is dead; aliases Q buf)
__global__ void k_transpose_h(const half_t* __restrict__ h16,
                              half_t* __restrict__ ht) {
  size_t i = (size_t)blockIdx.x * blockDim.x + threadIdx.x;
  int d = (int)(i % DD);
  size_t bs = i / DD;
  int s = (int)(bs % SS);
  int b = (int)(bs / SS);
  ht[((size_t)b * DD + d) * SS + s] = h16[i];
}

// ---- stage 1: Q = h @ W  (32x32 tile / wave) ------------------------------
// grid: (B*S/32)=512 row tiles x (800/32)=25 col tiles
__global__ void k_gemm_q(const half_t* __restrict__ h16,
                         const half_t* __restrict__ Wt,
                         half_t* __restrict__ Q) {
  const int wid = (blockIdx.x * blockDim.x + threadIdx.x) >> 5;
  const int lane = threadIdx.x & 31;
  const int mt = wid / 25, ng = wid % 25;
  const size_t row0 = (size_t)mt * 32;
  const int n0 = ng * 32;
  v8f c[2][2] = {};
#pragma unroll 1
  for (int k0 = 0; k0 < DD; k0 += 32) {
    v16h a0 = load_a_frag(h16 + row0 * DD + k0, DD, lane);
    v16h a1 = load_a_frag(h16 + (row0 + 16) * DD + k0, DD, lane);
    v16h b0 = load_b_frag(Wt + (size_t)n0 * DD + k0, DD, lane);
    v16h b1 = load_b_frag(Wt + (size_t)(n0 + 16) * DD + k0, DD, lane);
    c[0][0] = wmma_f16(a0, b0, c[0][0]);
    c[0][1] = wmma_f16(a0, b1, c[0][1]);
    c[1][0] = wmma_f16(a1, b0, c[1][0]);
    c[1][1] = wmma_f16(a1, b1, c[1][1]);
  }
#pragma unroll
  for (int i = 0; i < 2; ++i)
#pragma unroll
    for (int j = 0; j < 2; ++j)
      store_c_f16(c[i][j], Q + (row0 + 16 * i) * DD + n0 + 16 * j, DD, lane);
}

// ---- stage 2: scores = Q @ h^T  (32x64 tile / wave, f16 out) --------------
// grid = 8 * (2048/32)=64 * (2048/64)=32 waves
__global__ void k_scores(const half_t* __restrict__ Q,
                         const half_t* __restrict__ h16,
                         half_t* __restrict__ attn) {
  const int wid = (blockIdx.x * blockDim.x + threadIdx.x) >> 5;
  const int lane = threadIdx.x & 31;
  const int nt = wid & 31;
  const int mt = (wid >> 5) & 63;
  const int b = wid >> 11;
  const size_t row0 = (size_t)mt * 32;
  const int t0 = nt * 64;
  const half_t* Qb = Q + (size_t)b * SS * DD;
  const half_t* hb = h16 + (size_t)b * SS * DD;
  v8f c[2][4] = {};
#pragma unroll 1
  for (int k0 = 0; k0 < DD; k0 += 32) {
    v16h a[2], bf[4];
#pragma unroll
    for (int i = 0; i < 2; ++i)
      a[i] = load_a_frag(Qb + (row0 + 16 * i) * DD + k0, DD, lane);
#pragma unroll
    for (int j = 0; j < 4; ++j)
      bf[j] = load_b_frag(hb + (size_t)(t0 + 16 * j) * DD + k0, DD, lane);
#pragma unroll
    for (int i = 0; i < 2; ++i)
#pragma unroll
      for (int j = 0; j < 4; ++j) c[i][j] = wmma_f16(a[i], bf[j], c[i][j]);
  }
  half_t* ab = attn + (size_t)b * SS * SS;
#pragma unroll
  for (int i = 0; i < 2; ++i)
#pragma unroll
    for (int j = 0; j < 4; ++j)
      store_c_f16(c[i][j], ab + (row0 + 16 * i) * SS + t0 + 16 * j, SS, lane);
}

// ---- stage 3: in-place row softmax (f32 math) -----------------------------
__global__ void k_softmax(half_t* __restrict__ attn) {
  const size_t row = blockIdx.x;  // B*S rows of length S
  half_t* p = attn + row * SS;
  const int tid = threadIdx.x;  // 256 threads * 8 elems = 2048
  __shared__ float red[256];
  float v[8];
  v8h x = *(const v8h*)(p + tid * 8);
  float mx = -3.0e38f;
#pragma unroll
  for (int j = 0; j < 8; ++j) { v[j] = (float)x[j]; mx = fmaxf(mx, v[j]); }
  red[tid] = mx;
  __syncthreads();
  for (int s = 128; s > 0; s >>= 1) {
    if (tid < s) red[tid] = fmaxf(red[tid], red[tid + s]);
    __syncthreads();
  }
  mx = red[0];
  __syncthreads();
  float sum = 0.f;
#pragma unroll
  for (int j = 0; j < 8; ++j) { v[j] = __expf(v[j] - mx); sum += v[j]; }
  red[tid] = sum;
  __syncthreads();
  for (int s = 128; s > 0; s >>= 1) {
    if (tid < s) red[tid] += red[tid + s];
    __syncthreads();
  }
  const float inv = 1.0f / red[0];
  v8h o;
#pragma unroll
  for (int j = 0; j < 8; ++j) o[j] = (half_t)(v[j] * inv);
  *(v8h*)(p + tid * 8) = o;
}

// ---- stage 4: out = attn @ h  (64x32 tile / wave, f32 out) ----------------
// K = S = 2048 (64 steps); grid = 8 * (2048/64)=32 * (800/32)=25 waves
__global__ void k_out(const half_t* __restrict__ attn,
                      const half_t* __restrict__ ht,
                      float* __restrict__ out) {
  const int wid = (blockIdx.x * blockDim.x + threadIdx.x) >> 5;
  const int lane = threadIdx.x & 31;
  const int dg = wid % 25;
  const int rest = wid / 25;
  const int mt = rest & 31;
  const int b = rest >> 5;
  const size_t row0 = (size_t)mt * 64;
  const int d0 = dg * 32;
  const half_t* Ab = attn + (size_t)b * SS * SS;
  const half_t* Bb = ht + (size_t)b * DD * SS;  // [d][s]: K-contiguous per d
  v8f c[4][2] = {};
#pragma unroll 1
  for (int k0 = 0; k0 < SS; k0 += 32) {
    v16h a[4], bf[2];
#pragma unroll
    for (int i = 0; i < 4; ++i)
      a[i] = load_a_frag(Ab + (row0 + 16 * i) * SS + k0, SS, lane);
#pragma unroll
    for (int j = 0; j < 2; ++j)
      bf[j] = load_b_frag(Bb + (size_t)(d0 + 16 * j) * SS + k0, SS, lane);
#pragma unroll
    for (int i = 0; i < 4; ++i)
#pragma unroll
      for (int j = 0; j < 2; ++j) c[i][j] = wmma_f16(a[i], bf[j], c[i][j]);
  }
  float* ob = out + (size_t)b * SS * DD;
#pragma unroll
  for (int i = 0; i < 4; ++i)
#pragma unroll
    for (int j = 0; j < 2; ++j)
      store_c_f32(c[i][j], ob + (row0 + 16 * i) * DD + d0 + 16 * j, DD, lane);
}

// ---------------------------------------------------------------------------
extern "C" void kernel_launch(void* const* d_in, const int* in_sizes, int n_in,
                              void* d_out, int out_size, void* d_ws,
                              size_t ws_size, hipStream_t stream) {
  const float* h = (const float*)d_in[0];
  const float* W = (const float*)d_in[1];
  float* out = (float*)d_out;
  char* ws = (char*)d_ws;

  half_t* h16  = (half_t*)(ws);                 // 26,214,400 B
  half_t* Wt16 = (half_t*)(ws + 26214400);      //  1,280,000 B
  half_t* Q16  = (half_t*)(ws + 27494400);      // 26,214,400 B
  half_t* ht16 = (half_t*)(ws + 27494400);      // aliases Q (Q dead after scores)
  half_t* attn = (half_t*)(ws + 53708800);      // 67,108,864 B

  const int NH = BB * SS * DD;  // 13,107,200 (multiple of 256)

  k_convert_h<<<NH / 256, 256, 0, stream>>>(h, h16);
  k_convert_w<<<(DD * DD) / 256, 256, 0, stream>>>(W, Wt16);

  // Q = h @ W : 512*25 = 12800 waves, 4 waves/block
  k_gemm_q<<<(512 * 25) / 4, 128, 0, stream>>>(h16, Wt16, Q16);

  // scores = Q @ h^T : 8*64*32 = 16384 waves
  k_scores<<<(8 * 64 * 32) / 4, 128, 0, stream>>>(Q16, h16, attn);

  // softmax rows
  k_softmax<<<BB * SS, 256, 0, stream>>>(attn);

  // transpose h for B-operand of the final GEMM (into dead Q buffer)
  k_transpose_h<<<NH / 256, 256, 0, stream>>>(h16, ht16);

  // out = attn @ h : 8*32*25 = 6400 waves
  k_out<<<(8 * 32 * 25) / 4, 128, 0, stream>>>(attn, ht16, out);
}
